// Summarizer_76132590289451
// MI455X (gfx1250) — matrix operand order repaired and checked by reading, not compiled, not run on hardware
//
#include <hip/hip_runtime.h>
#include <hip/hip_bf16.h>

// ---------------------------------------------------------------------------
// Problem constants (from reference)
// ---------------------------------------------------------------------------
#define BB   32
#define NN_  32768
#define DD   128
#define TT   128
#define WIN_ 512
#define STEP_ 256
#define KK   2048
#define NE_  16
#define POS_ 33

typedef __attribute__((ext_vector_type(16))) _Float16 v16h;
typedef __attribute__((ext_vector_type(8)))  float    v8f;

union Frag16 {
    v16h v;
    unsigned u[8];
    _Float16 h[16];
};

__device__ __forceinline__ v8f wmma_f16(v16h a, v16h b, v8f c) {
    return __builtin_amdgcn_wmma_f32_16x16x32_f16(
        /*neg_a=*/false, a, /*neg_b=*/false, b,
        /*c_mod=*/(short)0, c, /*reuse_a=*/false, /*reuse_b=*/false);
}

// A-matrix fragment (16x32 f16), row-major source: row = lane&15 (+rowBase),
// element e of v16h has K = 16g + e; pairs contiguous -> 8 dwords.
__device__ __forceinline__ v16h load_a_frag(const _Float16* __restrict__ p,
                                            int row, int rowStride, int k0, int g) {
    Frag16 f;
    const _Float16* q = p + (long)row * rowStride + k0;
#pragma unroll
    for (int j = 0; j < 8; ++j) {
        int kb = 16 * g + 2 * j;
        f.u[j] = *reinterpret_cast<const unsigned*>(q + kb);
    }
    return f.v;
}

// B-matrix fragment (32x16 f16) from a K-contiguous ("[col][k]") layout:
// element e has K = k0 + 16g + e, column = lane&15 -> 16 contiguous halves.
__device__ __forceinline__ v16h load_b_frag_kcontig(const _Float16* __restrict__ p,
                                                    long col, int colStride, int k0, int g) {
    Frag16 f;
    const _Float16* q = p + col * colStride + k0 + 16 * g;
#pragma unroll
    for (int j = 0; j < 8; ++j)
        f.u[j] = *reinterpret_cast<const unsigned*>(q + 2 * j);
    return f.v;
}

// ---------------------------------------------------------------------------
// Utility kernels
// ---------------------------------------------------------------------------
__global__ void k_fill(float* __restrict__ p, long n, float v) {
    long i = (long)blockIdx.x * blockDim.x + threadIdx.x;
    if (i < n) p[i] = v;
}

__global__ void k_f32to16(const float* __restrict__ in, _Float16* __restrict__ out, long n) {
    long i = (long)blockIdx.x * blockDim.x + threadIdx.x;
    if (i < n) out[i] = (_Float16)in[i];
}

// gen_w (D, N) f32  ->  genwT (N, D) f16
__global__ void k_genwT(const float* __restrict__ gw, _Float16* __restrict__ out, long n) {
    long i = (long)blockIdx.x * blockDim.x + threadIdx.x;
    if (i < n) {
        int  d  = (int)(i & (DD - 1));
        long nc = i >> 7;
        out[i] = (_Float16)gw[(long)d * NN_ + nc];
    }
}

// ---------------------------------------------------------------------------
// Kernel 1: fused filterbank conv (implicit GEMM, WMMA) + |.| + avg-pool.
// Block (h, b): 256 spec positions p in [256h, 256h+256), all 128 filters.
// The |spec| sum of this half-block feeds frames h and h+1.
// ---------------------------------------------------------------------------
__global__ __launch_bounds__(256) void k_fbconv(const float* __restrict__ x,
                                                const _Float16* __restrict__ fbw,
                                                float* __restrict__ pooled) {
    const int h = blockIdx.x;   // 0..127
    const int b = blockIdx.y;

    __shared__ _Float16 xs[768];          // x[256h-256 .. 256h+511]
    const int x0 = h * 256 - 256;
    for (int i = threadIdx.x; i < 768; i += 256) {
        int gi = x0 + i;
        float v = (gi >= 0 && gi < NN_) ? x[(long)b * NN_ + gi] : 0.f;
        xs[i] = (_Float16)v;
    }
    __syncthreads();

    const int lane = threadIdx.x & 31;
    const int wave = threadIdx.x >> 5;    // 8 waves -> 16 filters each
    const int g    = lane >> 4;
    const int nn   = lane & 15;
    const int fRow = wave * 16 + nn;

    float sums[8] = {0.f, 0.f, 0.f, 0.f, 0.f, 0.f, 0.f, 0.f};
    const v8f zero = {};

    for (int ptc = 0; ptc < 16; ptc += 4) {       // 4 position tiles at a time
        v8f acc0 = zero, acc1 = zero, acc2 = zero, acc3 = zero;
        for (int k0 = 0; k0 < WIN_; k0 += 32) {
            v16h a = load_a_frag(fbw, fRow, WIN_, k0, g);
#pragma unroll
            for (int q = 0; q < 4; ++q) {
                Frag16 bf;
                const int base = (ptc + q) * 16 + nn + k0 + 16 * g;
#pragma unroll
                for (int e = 0; e < 16; ++e) bf.h[e] = xs[base + e];
                v8f* ap = (q == 0) ? &acc0 : (q == 1) ? &acc1 : (q == 2) ? &acc2 : &acc3;
                *ap = wmma_f16(a, bf.v, *ap);
            }
        }
#pragma unroll
        for (int r = 0; r < 8; ++r)
            sums[r] += fabsf(acc0[r]) + fabsf(acc1[r]) + fabsf(acc2[r]) + fabsf(acc3[r]);
    }

    // reduce over the 16 columns held by lanes nn=0..15 of each half-wave
#pragma unroll
    for (int off = 1; off < 16; off <<= 1)
#pragma unroll
        for (int r = 0; r < 8; ++r) sums[r] += __shfl_xor(sums[r], off, 32);

    if (nn == 0) {
#pragma unroll
        for (int r = 0; r < 8; ++r) {
            const int f = wave * 16 + r + 8 * g;
            const float v = sums[r] * (1.f / (float)WIN_);
            atomicAdd(&pooled[((long)b * DD + f) * TT + h], v);
            if (h + 1 < TT) atomicAdd(&pooled[((long)b * DD + f) * TT + h + 1], v);
        }
    }
}

// ---------------------------------------------------------------------------
// std over (channel,time) per example -> scale = 1/(std+1e-8)
// ---------------------------------------------------------------------------
__global__ __launch_bounds__(256) void k_std(const float* __restrict__ t,
                                             float* __restrict__ scale, int per) {
    const int b = blockIdx.x;
    __shared__ float s1[256], s2[256];
    float a = 0.f, q = 0.f;
    for (int i = threadIdx.x; i < per; i += 256) {
        float v = t[(long)b * per + i];
        a += v; q += v * v;
    }
    s1[threadIdx.x] = a; s2[threadIdx.x] = q;
    __syncthreads();
    for (int off = 128; off > 0; off >>= 1) {
        if (threadIdx.x < off) {
            s1[threadIdx.x] += s1[threadIdx.x + off];
            s2[threadIdx.x] += s2[threadIdx.x + off];
        }
        __syncthreads();
    }
    if (threadIdx.x == 0) {
        float mean = s1[0] / per;
        float var  = s2[0] / per - mean * mean;
        if (var < 0.f) var = 0.f;
        scale[b] = 1.f / (sqrtf(var) + 1e-8f);
    }
}

// ---------------------------------------------------------------------------
// pos-encode + 1x1 reduce conv:  h[b,d,t]
// ---------------------------------------------------------------------------
__global__ __launch_bounds__(128) void k_reduce(const float* __restrict__ pooled,
                                                const float* __restrict__ scale1,
                                                const float* __restrict__ rw,
                                                const float* __restrict__ rb,
                                                float* __restrict__ out) {
    const int d = blockIdx.x, b = blockIdx.y, t = threadIdx.x;
    const float sc = scale1[b];
    const float* w  = rw + (long)d * (DD + POS_);
    const float* pb = pooled + (long)b * DD * TT + t;
    float acc = rb[d];
    for (int c = 0; c < DD; ++c) acc += w[c] * (pb[c * TT] * sc);
    const float pos = -1.f + 2.f * (float)t / 127.f;
    acc += w[DD] * pos;
    float f = 3.14159265358979323846f;
#pragma unroll
    for (int i = 0; i < 16; ++i) {
        acc += w[DD + 1 + i] * sinf(pos * f) + w[DD + 17 + i] * cosf(pos * f);
        f *= 2.f;
    }
    out[((long)b * DD + d) * TT + t] = acc;
}

// ---------------------------------------------------------------------------
// one dilated residual leaky-relu block
// ---------------------------------------------------------------------------
__global__ __launch_bounds__(128) void k_dil(const float* __restrict__ hin,
                                             const float* __restrict__ w,
                                             const float* __restrict__ bias,
                                             float* __restrict__ hout, int dil) {
    const int d = blockIdx.x, b = blockIdx.y, t = threadIdx.x;
    const float* wd = w + (long)d * DD * 3;
    const float* hb = hin + (long)b * DD * TT;
    float y = bias[d];
    const int tm = t - dil, tp = t + dil;
    const bool okm = tm >= 0, okp = tp < TT;
    for (int c = 0; c < DD; ++c) {
        const float* hc = hb + c * TT;
        const float* wc = wd + c * 3;
        float a0 = okm ? hc[tm] : 0.f;
        float a2 = okp ? hc[tp] : 0.f;
        y += wc[0] * a0 + wc[1] * hc[t] + wc[2] * a2;
    }
    float act = y > 0.f ? y : 0.2f * y;
    hout[((long)b * DD + d) * TT + t] = hb[d * TT + t] + act;
}

// h (B,D,T) f32 * scale2[b]  ->  hN16 (B,T,D) f16   (K-contiguous for WMMA B)
__global__ void k_h16(const float* __restrict__ h, const float* __restrict__ scale2,
                      _Float16* __restrict__ out, long n) {
    long o = (long)blockIdx.x * blockDim.x + threadIdx.x;
    if (o >= n) return;
    int d = (int)(o & 127), t = (int)((o >> 7) & 127), b = (int)(o >> 14);
    out[o] = (_Float16)(h[((long)b * DD + d) * TT + t] * scale2[b]);
}

// ---------------------------------------------------------------------------
// expand: raw[b,k,t] = sum_d hN[b,d,t] * ew[k,d]    (WMMA, per-b GEMM 2048x128x128)
// ---------------------------------------------------------------------------
__global__ __launch_bounds__(256) void k_expand(const _Float16* __restrict__ ew,
                                                const _Float16* __restrict__ hn,
                                                float* __restrict__ raw) {
    const int b    = blockIdx.y;
    const int lane = threadIdx.x & 31, wave = threadIdx.x >> 5;
    const int tile = blockIdx.x * 8 + wave;         // 1024 tiles/b
    const int mt = tile >> 3, nt = tile & 7;
    const int g = lane >> 4, nn = lane & 15;
    v8f acc = {};
    const _Float16* hb = hn + (long)b * TT * DD;
    for (int k0 = 0; k0 < DD; k0 += 32) {
        v16h a  = load_a_frag(ew, mt * 16 + nn, DD, k0, g);
        v16h bb = load_b_frag_kcontig(hb, nt * 16 + nn, DD, k0, g);
        acc = wmma_f16(a, bb, acc);
    }
    const int tcol = nt * 16 + nn;
#pragma unroll
    for (int r = 0; r < 8; ++r) {
        const int k = mt * 16 + r + 8 * g;
        raw[((long)b * KK + k) * TT + tcol] = acc[r];
    }
}

// ---------------------------------------------------------------------------
// softmax over K per (b,t)
// ---------------------------------------------------------------------------
__global__ __launch_bounds__(256) void k_softmax(const float* __restrict__ raw,
                                                 float* __restrict__ sm) {
    const int t = blockIdx.x, b = blockIdx.y;
    __shared__ float red[256];
    const float* col = raw + (long)b * KK * TT + t;
    float mx = -1e30f;
    for (int k = threadIdx.x; k < KK; k += 256) mx = fmaxf(mx, col[(long)k * TT]);
    red[threadIdx.x] = mx; __syncthreads();
    for (int off = 128; off > 0; off >>= 1) {
        if (threadIdx.x < off) red[threadIdx.x] = fmaxf(red[threadIdx.x], red[threadIdx.x + off]);
        __syncthreads();
    }
    mx = red[0]; __syncthreads();
    float sum = 0.f;
    for (int k = threadIdx.x; k < KK; k += 256) sum += expf(col[(long)k * TT] - mx);
    red[threadIdx.x] = sum; __syncthreads();
    for (int off = 128; off > 0; off >>= 1) {
        if (threadIdx.x < off) red[threadIdx.x] += red[threadIdx.x + off];
        __syncthreads();
    }
    const float inv = 1.f / red[0];
    float* smb = sm + (long)b * KK * TT + t;
    for (int k = threadIdx.x; k < KK; k += 256) smb[(long)k * TT] = expf(col[(long)k * TT] - mx) * inv;
}

// ---------------------------------------------------------------------------
// top-16 over flat (K*T) per example
// ---------------------------------------------------------------------------
__global__ __launch_bounds__(256) void k_topk(const float* __restrict__ sm,
                                              float* __restrict__ vals,
                                              int* __restrict__ idxo) {
    const int b = blockIdx.x;
    const float* p = sm + (long)b * (KK * TT);
    float lv[16]; int li[16];
#pragma unroll
    for (int r = 0; r < 16; ++r) { lv[r] = -1e30f; li[r] = 0x7fffffff; }
    for (int i = threadIdx.x; i < KK * TT; i += 256) {
        float v = p[i];
        if (v > lv[15] || (v == lv[15] && i < li[15])) {
            int pos = 15;
            while (pos > 0 && (v > lv[pos - 1] || (v == lv[pos - 1] && i < li[pos - 1]))) {
                lv[pos] = lv[pos - 1]; li[pos] = li[pos - 1]; --pos;
            }
            lv[pos] = v; li[pos] = i;
        }
    }
    __shared__ float cv[4096]; __shared__ int ci[4096];
    __shared__ float bv[256];  __shared__ int bi[256]; __shared__ int bs[256];
#pragma unroll
    for (int r = 0; r < 16; ++r) {
        cv[threadIdx.x * 16 + r] = lv[r];
        ci[threadIdx.x * 16 + r] = li[r];
    }
    __syncthreads();
    for (int round = 0; round < 16; ++round) {
        float mv = -1e30f; int mi = 0x7fffffff; int ms = 0;
#pragma unroll
        for (int r = 0; r < 16; ++r) {
            int s = threadIdx.x * 16 + r;
            float v = cv[s]; int ii = ci[s];
            if (v > mv || (v == mv && ii < mi)) { mv = v; mi = ii; ms = s; }
        }
        bv[threadIdx.x] = mv; bi[threadIdx.x] = mi; bs[threadIdx.x] = ms;
        __syncthreads();
        for (int off = 128; off > 0; off >>= 1) {
            if (threadIdx.x < off) {
                float v2 = bv[threadIdx.x + off]; int i2 = bi[threadIdx.x + off];
                if (v2 > bv[threadIdx.x] || (v2 == bv[threadIdx.x] && i2 < bi[threadIdx.x])) {
                    bv[threadIdx.x] = v2; bi[threadIdx.x] = i2; bs[threadIdx.x] = bs[threadIdx.x + off];
                }
            }
            __syncthreads();
        }
        if (threadIdx.x == 0) {
            vals[b * NE_ + round] = bv[0];
            idxo[b * NE_ + round] = bi[0];
            cv[bs[0]] = -1e30f;
        }
        __syncthreads();
    }
}

// ---------------------------------------------------------------------------
// events: oh16[e,d] = val_e * embed_w[d, chan_e]; scatter sparse values
// ---------------------------------------------------------------------------
__global__ __launch_bounds__(128) void k_events(const float* __restrict__ vals,
                                                const int* __restrict__ idxo,
                                                const float* __restrict__ embed,
                                                _Float16* __restrict__ oh16,
                                                float* __restrict__ sparse) {
    const int e = blockIdx.x, d = threadIdx.x;
    const float v = vals[e];
    const int fl = idxo[e];
    const int ch = fl >> 7;                 // / T
    oh16[(long)e * DD + d] = (_Float16)(v * embed[(long)d * KK + ch]);
    if (d == 0) {
        const int b = e >> 4;               // / NE
        sparse[(long)b * (KK * TT) + fl] = v;
    }
}

// ---------------------------------------------------------------------------
// generator: seg[e,n] = tanh(sum_d oh[e,d] * gen_w[d,n])   (WMMA 512x32768x128)
// ---------------------------------------------------------------------------
__global__ __launch_bounds__(256) void k_gen(const _Float16* __restrict__ oh16,
                                             const _Float16* __restrict__ gwT,
                                             _Float16* __restrict__ seg) {
    const int lane = threadIdx.x & 31, wave = threadIdx.x >> 5;
    const int tile = blockIdx.x * 8 + wave;     // 65536 tiles
    const int nt = tile & 2047, mt = tile >> 11;
    const int g = lane >> 4, nn = lane & 15;
    v8f acc = {};
    for (int k0 = 0; k0 < DD; k0 += 32) {
        v16h a  = load_a_frag(oh16, mt * 16 + nn, DD, k0, g);
        v16h bb = load_b_frag_kcontig(gwT, nt * 16 + nn, DD, k0, g);
        acc = wmma_f16(a, bb, acc);
    }
    const int j = nt * 16 + nn;
#pragma unroll
    for (int r = 0; r < 8; ++r) {
        const int ev = mt * 16 + r + 8 * g;
        seg[(long)ev * NN_ + j] = (_Float16)tanhf(acc[r]);
    }
}

// ---------------------------------------------------------------------------
// deterministic overlap-add gather into out[:, :N]
// ---------------------------------------------------------------------------
__global__ void k_final(const _Float16* __restrict__ seg, const int* __restrict__ idxo,
                        float* __restrict__ out, long n) {
    long o = (long)blockIdx.x * blockDim.x + threadIdx.x;
    if (o >= n) return;
    const int p = (int)(o & (NN_ - 1));
    const int b = (int)(o >> 15);
    float s = 0.f;
    for (int i = 0; i < NE_; ++i) {
        const int start = (idxo[b * NE_ + i] & (TT - 1)) * STEP_;
        const int j = p - start;
        if (j >= 0) s += (float)seg[(long)(b * NE_ + i) * NN_ + j];
    }
    out[o] = s;
}

// ---------------------------------------------------------------------------
// host launcher
// ---------------------------------------------------------------------------
extern "C" void kernel_launch(void* const* d_in, const int* in_sizes, int n_in,
                              void* d_out, int out_size, void* d_ws, size_t ws_size,
                              hipStream_t stream) {
    (void)in_sizes; (void)n_in; (void)out_size; (void)ws_size;

    const float* x        = (const float*)d_in[0];
    const float* fb_w     = (const float*)d_in[1];
    const float* reduce_w = (const float*)d_in[2];
    const float* reduce_b = (const float*)d_in[3];
    const float* dil_w    = (const float*)d_in[4];
    const float* dil_b    = (const float*)d_in[5];
    const float* expand_w = (const float*)d_in[6];
    const float* embed_w  = (const float*)d_in[7];
    const float* gen_w    = (const float*)d_in[8];

    float* out    = (float*)d_out;                          // (B,1,N)
    float* sparse = out + (size_t)BB * NN_;                 // (B,K,T)
    float* raw    = sparse + (size_t)BB * KK * TT;          // (B,K,T)

    // workspace partition (256B aligned slabs)
    char* wsp = (char*)d_ws;
    auto alloc = [&](size_t bytes) -> char* {
        char* p = wsp;
        wsp += (bytes + 255) & ~(size_t)255;
        return p;
    };
    _Float16* fbw16  = (_Float16*)alloc((size_t)DD * WIN_ * 2);        // 128K
    _Float16* expw16 = (_Float16*)alloc((size_t)KK * DD * 2);          // 512K
    _Float16* genwT  = (_Float16*)alloc((size_t)NN_ * DD * 2);         // 8M
    float*    pooled = (float*)  alloc((size_t)BB * DD * TT * 4);      // 2M
    float*    scale1 = (float*)  alloc(BB * 4);
    float*    scale2 = (float*)  alloc(BB * 4);
    float*    hA     = (float*)  alloc((size_t)BB * DD * TT * 4);      // 2M
    float*    hB     = (float*)  alloc((size_t)BB * DD * TT * 4);      // 2M
    _Float16* hN16   = (_Float16*)alloc((size_t)BB * TT * DD * 2);     // 1M
    float*    sm     = (float*)  alloc((size_t)BB * KK * TT * 4);      // 32M
    float*    vals   = (float*)  alloc((size_t)BB * NE_ * 4);
    int*      idxA   = (int*)    alloc((size_t)BB * NE_ * 4);
    _Float16* oh16   = (_Float16*)alloc((size_t)BB * NE_ * DD * 2);    // 128K
    _Float16* seg    = (_Float16*)alloc((size_t)BB * NE_ * NN_ * 2);   // 32M

    const long nPooled = (long)BB * DD * TT;
    const long nH16    = nPooled;
    const long nGw     = (long)NN_ * DD;
    const long nSparse = (long)BB * KK * TT;
    const long nOut    = (long)BB * NN_;

    // prep
    k_fill   <<<(int)((nPooled + 255) / 256), 256, 0, stream>>>(pooled, nPooled, 0.f);
    k_f32to16<<<(DD * WIN_ + 255) / 256, 256, 0, stream>>>(fb_w, fbw16, (long)DD * WIN_);
    k_f32to16<<<(KK * DD + 255) / 256, 256, 0, stream>>>(expand_w, expw16, (long)KK * DD);
    k_genwT  <<<(int)((nGw + 255) / 256), 256, 0, stream>>>(gen_w, genwT, nGw);

    // 1) filterbank conv + abs + pool (WMMA)
    k_fbconv<<<dim3(TT, BB), 256, 0, stream>>>(x, fbw16, pooled);

    // 2) norm1, reduce conv
    k_std   <<<BB, 256, 0, stream>>>(pooled, scale1, DD * TT);
    k_reduce<<<dim3(DD, BB), 128, 0, stream>>>(pooled, scale1, reduce_w, reduce_b, hA);

    // 3) dilated residual stack (ping-pong)
    const int dils[5] = {1, 3, 9, 27, 1};
    float* cur = hA; float* nxt = hB;
    for (int i = 0; i < 5; ++i) {
        k_dil<<<dim3(DD, BB), 128, 0, stream>>>(cur, dil_w + (size_t)i * DD * DD * 3,
                                                dil_b + (size_t)i * DD, nxt, dils[i]);
        float* tmp = cur; cur = nxt; nxt = tmp;
    }

    // 4) norm2 + f16 transpose
    k_std<<<BB, 256, 0, stream>>>(cur, scale2, DD * TT);
    k_h16<<<(int)((nH16 + 255) / 256), 256, 0, stream>>>(cur, scale2, hN16, nH16);

    // 5) expand (WMMA) -> raw output
    k_expand<<<dim3(128, BB), 256, 0, stream>>>(expw16, hN16, raw);

    // 6) softmax, sparse zero-fill, top-k, sparse scatter + event embed
    k_softmax<<<dim3(TT, BB), 256, 0, stream>>>(raw, sm);
    k_fill   <<<(int)((nSparse + 255) / 256), 256, 0, stream>>>(sparse, nSparse, 0.f);
    k_topk   <<<BB, 256, 0, stream>>>(sm, vals, idxA);
    k_events <<<BB * NE_, 128, 0, stream>>>(vals, idxA, embed_w, oh16, sparse);

    // 7) generator (WMMA + tanh), deterministic overlap-add gather
    k_gen  <<<8192, 256, 0, stream>>>(oh16, genwT, seg);
    k_final<<<(int)((nOut + 255) / 256), 256, 0, stream>>>(seg, idxA, out, nOut);
}
